// LSTMModel_52828097741376
// MI455X (gfx1250) — compile-verified, matrix operand order checked by reading, not run on hardware
//
#include <hip/hip_runtime.h>

typedef __attribute__((ext_vector_type(16))) _Float16 v16h;
typedef __attribute__((ext_vector_type(8)))  float    v8f;

#define HID 64
#define G4H 256
#define ENC_T 20
#define DEC_T 30
#define ROWS_PER_WAVE 16
#define WAVES_PER_WG 4
#define ROWS_PER_WG (ROWS_PER_WAVE * WAVES_PER_WG)

#define LOG2E  1.4426950408889634f
#define LOG2E2 2.8853900817779268f

// Branch-free hardware transcendentals (v_exp_f32 is base-2 on AMD).
__device__ __forceinline__ float fast_sig(float x) {
  return __builtin_amdgcn_rcpf(1.0f + __builtin_amdgcn_exp2f(x * -LOG2E));
}
__device__ __forceinline__ float fast_tanh(float x) {
  return 1.0f - 2.0f * __builtin_amdgcn_rcpf(1.0f + __builtin_amdgcn_exp2f(x * LOG2E2));
}
// Pre-scaled-bias variants: b already multiplied by -LOG2E / LOG2E2.
__device__ __forceinline__ float fast_sig_b(float x, float bscaled) {
  return __builtin_amdgcn_rcpf(1.0f + __builtin_amdgcn_exp2f(x * -LOG2E + bscaled));
}
__device__ __forceinline__ float fast_tanh_b(float x, float bscaled) {
  return 1.0f - 2.0f * __builtin_amdgcn_rcpf(1.0f + __builtin_amdgcn_exp2f(x * LOG2E2 + bscaled));
}

// Wave-local LDS ordering point: sH/sX are wave-private and same-wave DS ops
// complete in order (ISA 7.3) -- only compiler reordering must be blocked.
__device__ __forceinline__ void wave_lds_sync() {
  __builtin_amdgcn_fence(__ATOMIC_ACQ_REL, "workgroup");
  __builtin_amdgcn_wave_barrier();
}

// Load a 16x32 f16 WMMA fragment (A or B role) from an LDS tile stored
// row-major with 64 halves per row (ISA 7.12.2 16-bit layout).
__device__ __forceinline__ v16h load_frag(const _Float16* base, int rowHalf, int chunk, int hi) {
  union { v16h v; unsigned u[8]; } r;
  const unsigned* p = (const unsigned*)base;
  const int rb = rowHalf + 32 * chunk + 8 * hi;   // always even
#pragma unroll
  for (int d = 0; d < 8; ++d) {
    const int k = (d < 4) ? (2 * d) : (2 * d + 8);
    r.u[d] = p[(rb + k) >> 1];                    // pairs merge into ds_load_b128
  }
  return r.v;
}

__global__ __launch_bounds__(128) void lstm_seq2seq_wmma(
    const float* __restrict__ traj,
    const float* __restrict__ Wih_e, const float* __restrict__ Whh_e,
    const float* __restrict__ bih_e, const float* __restrict__ bhh_e,
    const float* __restrict__ Wih_d, const float* __restrict__ Whh_d,
    const float* __restrict__ bih_d, const float* __restrict__ bhh_d,
    const float* __restrict__ Wemb,  const float* __restrict__ bemb,
    float* __restrict__ out) {
  __shared__ _Float16 sW[G4H * HID];                          // 32 KB, enc->dec reuse
  __shared__ float sB_enc[G4H], sB_dec[G4H];                  // fused biases
  __shared__ float sWi0[G4H], sWi1[G4H];                      // encoder input cols
  __shared__ float sWem[2 * HID];                             // output projection
  __shared__ _Float16 sH[WAVES_PER_WG][ROWS_PER_WAVE * HID];  // per-wave h (f16)
  __shared__ float sX[WAVES_PER_WG][ROWS_PER_WAVE * ENC_T * 2]; // per-wave traj slab

  const int tid  = threadIdx.x;
  const int w    = tid >> 5;
  const int lane = tid & 31;
  const int hi   = lane >> 4;   // lane half select
  const int lr   = lane & 15;   // lane within half
  const int rowBase = blockIdx.x * ROWS_PER_WG + w * ROWS_PER_WAVE;

  // ---- one-time prep ----
  for (int i = tid; i < G4H * HID; i += 128) sW[i] = (_Float16)Whh_e[i];
  for (int i = tid; i < G4H; i += 128) {
    sB_enc[i] = bih_e[i] + bhh_e[i];
    sB_dec[i] = bih_d[i] + bhh_d[i];
    sWi0[i]   = Wih_e[2 * i + 0];
    sWi1[i]   = Wih_e[2 * i + 1];
  }
  if (tid < 2 * HID) sWem[tid] = Wemb[tid];
  // stage this wave's 16x(20x2) trajectory slab with coalesced float4 loads
  {
    const float4* g4 = (const float4*)(traj + (size_t)rowBase * (ENC_T * 2));
    float4* l4 = (float4*)sX[w];
#pragma unroll
    for (int it = 0; it < (ROWS_PER_WAVE * ENC_T * 2) / (4 * 32); ++it)
      l4[lane + it * 32] = g4[lane + it * 32];
  }
  // zero this wave's h staging tile (1024 halves = 512 dwords)
  {
    unsigned* hz = (unsigned*)sH[w];
#pragma unroll
    for (int d = 0; d < 16; ++d) hz[lane * 16 + d] = 0u;
  }
  v8f cregs[4];  // c state in WMMA C/D layout: cregs[jt][v] = c(row v+8*hi, col jt*16+lr)
#pragma unroll
  for (int jt = 0; jt < 4; ++jt)
#pragma unroll
    for (int v = 0; v < 8; ++v) cregs[jt][v] = 0.0f;
  __syncthreads();

  // per-lane decoder biases, pre-scaled into the exp2 argument domain
  float bdi[4], bdf[4], bdg[4], bdo[4];
#pragma unroll
  for (int jt = 0; jt < 4; ++jt) {
    bdi[jt] = -LOG2E  * sB_dec[(0 * 4 + jt) * 16 + lr];
    bdf[jt] = -LOG2E  * sB_dec[(1 * 4 + jt) * 16 + lr];
    bdg[jt] =  LOG2E2 * sB_dec[(2 * 4 + jt) * 16 + lr];
    bdo[jt] = -LOG2E  * sB_dec[(3 * 4 + jt) * 16 + lr];
  }

  // ------------------------------ encoder ------------------------------
  for (int t = 0; t < ENC_T; ++t) {
    wave_lds_sync();
    float xr0[8], xr1[8];
#pragma unroll
    for (int v = 0; v < 8; ++v) {
      const int m = v + 8 * hi;
      xr0[v] = sX[w][m * (ENC_T * 2) + t * 2 + 0];
      xr1[v] = sX[w][m * (ENC_T * 2) + t * 2 + 1];
    }
    const v16h a0 = load_frag(sH[w], lr * HID, 0, hi);
    const v16h a1 = load_frag(sH[w], lr * HID, 1, hi);
    wave_lds_sync();
#pragma unroll
    for (int jt = 0; jt < 4; ++jt) {
      v8f acc[4];
#pragma unroll
      for (int g = 0; g < 4; ++g) {               // gate order i,f,g,o
        const int n = (g * 4 + jt) * 16 + lr;     // global gate column
        v8f ci;
#pragma unroll
        for (int v = 0; v < 8; ++v)
          ci[v] = sB_enc[n] + xr0[v] * sWi0[n] + xr1[v] * sWi1[n];
        const v16h b0 = load_frag(sW, n * HID, 0, hi);
        const v16h b1 = load_frag(sW, n * HID, 1, hi);
        v8f d = __builtin_amdgcn_wmma_f32_16x16x32_f16(false, a0, false, b0,
                                                       (short)0, ci, false, false);
        d = __builtin_amdgcn_wmma_f32_16x16x32_f16(false, a1, false, b1,
                                                   (short)0, d, false, false);
        acc[g] = d;
      }
#pragma unroll
      for (int v = 0; v < 8; ++v) {
        const float cn = fast_sig(acc[1][v]) * cregs[jt][v] +
                         fast_sig(acc[0][v]) * fast_tanh(acc[2][v]);
        const float hn = fast_sig(acc[3][v]) * fast_tanh(cn);
        cregs[jt][v] = cn;
        sH[w][(v + 8 * hi) * HID + jt * 16 + lr] = (_Float16)hn;
      }
    }
  }

  // ---- swap weights: decoder uses x==h, so fold Wih+Whh into one matrix ----
  __syncthreads();
  for (int i = tid; i < G4H * HID; i += 128) sW[i] = (_Float16)(Wih_d[i] + Whh_d[i]);
  __syncthreads();

  // ------------------------------ decoder ------------------------------
  const float bout = bemb[hi];
  const v8f zeroC = {};
  for (int t = 0; t < DEC_T; ++t) {
    wave_lds_sync();
    const v16h a0 = load_frag(sH[w], lr * HID, 0, hi);
    const v16h a1 = load_frag(sH[w], lr * HID, 1, hi);
    wave_lds_sync();
#pragma unroll
    for (int jt = 0; jt < 4; ++jt) {
      v8f acc[4];
#pragma unroll
      for (int g = 0; g < 4; ++g) {
        const int n = (g * 4 + jt) * 16 + lr;
        const v16h b0 = load_frag(sW, n * HID, 0, hi);
        const v16h b1 = load_frag(sW, n * HID, 1, hi);
        v8f d = __builtin_amdgcn_wmma_f32_16x16x32_f16(false, a0, false, b0,
                                                       (short)0, zeroC, false, false);
        d = __builtin_amdgcn_wmma_f32_16x16x32_f16(false, a1, false, b1,
                                                   (short)0, d, false, false);
        acc[g] = d;
      }
#pragma unroll
      for (int v = 0; v < 8; ++v) {
        // biases folded into the exp2-argument FMA (no broadcast C-init needed)
        const float cn = fast_sig_b(acc[1][v], bdf[jt]) * cregs[jt][v] +
                         fast_sig_b(acc[0][v], bdi[jt]) * fast_tanh_b(acc[2][v], bdg[jt]);
        const float hn = fast_sig_b(acc[3][v], bdo[jt]) * fast_tanh(cn);
        cregs[jt][v] = cn;
        sH[w][(v + 8 * hi) * HID + jt * 16 + lr] = (_Float16)hn;
      }
    }
    wave_lds_sync();
    // output projection: lane lr -> batch row, lane half -> component
    float s = bout;
    const unsigned* ph = (const unsigned*)(sH[w] + lr * HID);
#pragma unroll
    for (int jj = 0; jj < 32; ++jj) {
      union { unsigned u; _Float16 h[2]; } cv;
      cv.u = ph[jj];
      s += (float)cv.h[0] * sWem[hi * HID + 2 * jj + 0] +
           (float)cv.h[1] * sWem[hi * HID + 2 * jj + 1];
    }
    out[(size_t)(rowBase + lr) * (DEC_T * 2) + t * 2 + hi] = s;
  }
}

extern "C" void kernel_launch(void* const* d_in, const int* in_sizes, int n_in,
                              void* d_out, int out_size, void* d_ws, size_t ws_size,
                              hipStream_t stream) {
  (void)n_in; (void)out_size; (void)d_ws; (void)ws_size;
  const int B = in_sizes[0] / (ENC_T * 2);
  const int blocks = B / ROWS_PER_WG;
  lstm_seq2seq_wmma<<<blocks, 128, 0, stream>>>(
      (const float*)d_in[0],
      (const float*)d_in[1], (const float*)d_in[2],
      (const float*)d_in[3], (const float*)d_in[4],
      (const float*)d_in[5], (const float*)d_in[6],
      (const float*)d_in[7], (const float*)d_in[8],
      (const float*)d_in[9], (const float*)d_in[10],
      (float*)d_out);
}
